// LinearAttention_73237782331959
// MI455X (gfx1250) — matrix operand order repaired and checked by reading, not compiled, not run on hardware
//
#include <hip/hip_runtime.h>

// ---------------------------------------------------------------------------
// MI455X (gfx1250) linear attention.
// All GEMMs via v_wmma_f32_16x16x32_bf16 (wave32, 16x16x32 tiles).
// Softmax / RMS-norm statistics kept in f32.
// ---------------------------------------------------------------------------

typedef __bf16 bf16;
typedef __attribute__((ext_vector_type(16))) __bf16 bf16x16;
typedef __attribute__((ext_vector_type(8)))  __bf16 bf16x8;
typedef __attribute__((ext_vector_type(2)))  __bf16 bf16x2;
typedef __attribute__((ext_vector_type(8)))  float  f32x8;
typedef int v4i __attribute__((vector_size(16)));   // matches async builtin pointee

#define NPIX   16384     // 128*128
#define CDIM   512
#define HEADS  8
#define DHEAD  64
#define LDST   40        // LDS row stride (bf16 elems): 80B, 16B-aligned chunks
#define LDO    72        // LDS row stride for K=64 tiles: 144B

#if __has_builtin(__builtin_amdgcn_global_load_async_to_lds_b128) && \
    __has_builtin(__builtin_amdgcn_s_wait_asynccnt)
#define USE_ASYNC_LDS 1
#define AS1 __attribute__((address_space(1)))
#define AS3 __attribute__((address_space(3)))
#endif

// ---- WMMA helpers (CDNA5 VGPR layouts, ISA 7.12.2) -------------------------

static __device__ inline f32x8 wmma_bf16(bf16x16 a, bf16x16 b, f32x8 c) {
  return __builtin_amdgcn_wmma_f32_16x16x32_bf16(
      /*neg_a=*/false, a, /*neg_b=*/false, b,
      /*c_mod=*/(short)0, c, /*reuse_a=*/false, /*reuse_b=*/false);
}

// A fragment (16x32, 16-bit): lane<16 -> K {0..7,16..23}, lane>=16 -> +8.
static __device__ inline bf16x16 frag_a(const bf16* tile, int stride,
                                        int m_base, int lane, int kk) {
  const int r  = m_base + (lane & 15);
  const int kb = (lane & 16) ? 8 : 0;
  const bf16* p = tile + r * stride + kk + kb;
  bf16x8 lo = *(const bf16x8*)(p);
  bf16x8 hi = *(const bf16x8*)(p + 16);
  bf16x16 f;
#pragma unroll
  for (int i = 0; i < 8; ++i) { f[i] = lo[i]; f[i + 8] = hi[i]; }
  return f;
}

// B fragment (32x16, 16-bit): col = lane&15, lane<16 -> K {0..7,16..23}, lane>=16 -> +8.
// Tile stored transposed in LDS: [n][k].
static __device__ inline bf16x16 frag_b(const bf16* tile, int stride,
                                        int n_base, int lane, int kk) {
  const int c  = n_base + (lane & 15);
  const int kb = (lane & 16) ? 8 : 0;
  const bf16* p = tile + c * stride + kk + kb;
  bf16x8 lo = *(const bf16x8*)(p);
  bf16x8 hi = *(const bf16x8*)(p + 16);
  bf16x16 f;
#pragma unroll
  for (int i = 0; i < 8; ++i) { f[i] = lo[i]; f[i + 8] = hi[i]; }
  return f;
}

// ---- Generic 128x128-tile GEMM: C[M,N] = A[M,K] * B[K,N] -------------------
// A,B bf16 row-major. Output f32 (+bias) or bf16. Double-buffered LDS.
__global__ void __launch_bounds__(256) gemm128_bf16(
    const bf16* __restrict__ A, const bf16* __restrict__ B,
    int M, int N, int K,
    float* __restrict__ Cf, const float* __restrict__ bias,
    bf16* __restrict__ Cb)
{
  __shared__ bf16 sA[2][128 * LDST];
  __shared__ bf16 sB[2][128 * LDST];

  const int t    = threadIdx.x;
  const int lane = t & 31;
  const int wid  = t >> 5;
  const int mblk = blockIdx.y * 128;
  const int nblk = blockIdx.x * 128;
  const int wm   = (wid >> 1) * 32;   // wave tile: 32(m) x 64(n)
  const int wn   = (wid & 1) * 64;

  f32x8 acc[2][4];
#pragma unroll
  for (int mt = 0; mt < 2; ++mt)
#pragma unroll
    for (int nt = 0; nt < 4; ++nt)
#pragma unroll
      for (int r = 0; r < 8; ++r) acc[mt][nt][r] = 0.0f;

  const int a_m  = t >> 1;          // 0..127
  const int a_k  = (t & 1) * 16;    // 0 / 16
  const int b_kp = (t >> 4) * 2;    // 0..30 (even K pair)
  const int b_n0 = (t & 15) * 8;    // 0..120

  uint4 av0, av1, bv0, bv1;
  // prologue: fetch tile 0 into registers and stage into buffer 0
  {
    const bf16* ga = A + (size_t)(mblk + a_m) * K + a_k;
    av0 = ((const uint4*)ga)[0];
    av1 = ((const uint4*)ga)[1];
    const bf16* gb = B + (size_t)b_kp * N + nblk + b_n0;
    bv0 = *(const uint4*)gb;
    bv1 = *(const uint4*)(gb + N);
  }
  {
    *(uint4*)&sA[0][a_m * LDST + a_k]     = av0;
    *(uint4*)&sA[0][a_m * LDST + a_k + 8] = av1;
    bf16 r0[8], r1[8];
    *(uint4*)r0 = bv0; *(uint4*)r1 = bv1;
#pragma unroll
    for (int i = 0; i < 8; ++i) {
      bf16x2 pr; pr[0] = r0[i]; pr[1] = r1[i];
      *(bf16x2*)&sB[0][(b_n0 + i) * LDST + b_kp] = pr;
    }
  }
  __syncthreads();

  int buf = 0;
  for (int k0 = 0; k0 < K; k0 += 32, buf ^= 1) {
    const bool more = (k0 + 32) < K;
    if (more) {  // fetch next tile while current computes
      const bf16* ga = A + (size_t)(mblk + a_m) * K + k0 + 32 + a_k;
      av0 = ((const uint4*)ga)[0];
      av1 = ((const uint4*)ga)[1];
      const bf16* gb = B + (size_t)(k0 + 32 + b_kp) * N + nblk + b_n0;
      bv0 = *(const uint4*)gb;
      bv1 = *(const uint4*)(gb + N);
      if (k0 + 64 < K) {  // gfx1250 global_prefetch_b8, near-cache scope
        __builtin_prefetch(A + (size_t)(mblk + a_m) * K + k0 + 64 + a_k, 0, 3);
        __builtin_prefetch(B + (size_t)(k0 + 64 + b_kp) * N + nblk + b_n0, 0, 3);
      }
    }

    bf16x16 af[2], bfr[4];
#pragma unroll
    for (int mt = 0; mt < 2; ++mt) af[mt]  = frag_a(sA[buf], LDST, wm + mt * 16, lane, 0);
#pragma unroll
    for (int nt = 0; nt < 4; ++nt) bfr[nt] = frag_b(sB[buf], LDST, wn + nt * 16, lane, 0);
#pragma unroll
    for (int mt = 0; mt < 2; ++mt)
#pragma unroll
      for (int nt = 0; nt < 4; ++nt)
        acc[mt][nt] = wmma_bf16(af[mt], bfr[nt], acc[mt][nt]);

    if (more) {  // stage next tile into the other buffer
      const int nb = buf ^ 1;
      *(uint4*)&sA[nb][a_m * LDST + a_k]     = av0;
      *(uint4*)&sA[nb][a_m * LDST + a_k + 8] = av1;
      bf16 r0[8], r1[8];
      *(uint4*)r0 = bv0; *(uint4*)r1 = bv1;
#pragma unroll
      for (int i = 0; i < 8; ++i) {
        bf16x2 pr; pr[0] = r0[i]; pr[1] = r1[i];
        *(bf16x2*)&sB[nb][(b_n0 + i) * LDST + b_kp] = pr;
      }
    }
    __syncthreads();
  }

  const int mhalf = (lane & 16) ? 8 : 0;
#pragma unroll
  for (int mt = 0; mt < 2; ++mt)
#pragma unroll
    for (int nt = 0; nt < 4; ++nt)
#pragma unroll
      for (int r = 0; r < 8; ++r) {
        const int m = mblk + wm + mt * 16 + r + mhalf;
        const int n = nblk + wn + nt * 16 + (lane & 15);
        float v = acc[mt][nt][r];
        if (bias) v += bias[m];
        if (Cf) Cf[(size_t)m * N + n] = v;
        else    Cb[(size_t)m * N + n] = (bf16)v;
      }
}

// ---- context GEMM (per head, split-K, atomic f32 accumulate) ---------------
// ctx[h][d][e] += sum_n kb[h][d][n] * vb[h][e][n]   (K-chunk = 1024 per block)
__global__ void __launch_bounds__(256) ctx_gemm_k(
    const bf16* __restrict__ kbuf, const bf16* __restrict__ vbuf,
    float* __restrict__ ctx)
{
  __shared__ bf16 sA[64 * LDST];
  __shared__ bf16 sB[64 * LDST];
  const int h    = blockIdx.y;
  const int kc0  = blockIdx.x * 1024;
  const int t    = threadIdx.x;
  const int lane = t & 31;
  const int wid  = t >> 5;
  const int wm   = (wid >> 1) * 16;   // wave tile 16(d) x 32(e)
  const int wn   = (wid & 1) * 32;

  f32x8 acc[2];
#pragma unroll
  for (int nt = 0; nt < 2; ++nt)
#pragma unroll
    for (int r = 0; r < 8; ++r) acc[nt][r] = 0.0f;

  const int l_row = t >> 2;        // 0..63
  const int l_k8  = (t & 3) * 8;   // 0..24

  for (int kk = 0; kk < 1024; kk += 32) {
    const bf16* ga = kbuf + (size_t)(h * 64 + l_row) * NPIX + kc0 + kk + l_k8;
    const bf16* gb = vbuf + (size_t)(h * 64 + l_row) * NPIX + kc0 + kk + l_k8;
#ifdef USE_ASYNC_LDS
    // gfx1250 async DMA: global -> LDS without touching VGPRs (ASYNCcnt)
    __builtin_amdgcn_global_load_async_to_lds_b128(
        (AS1 v4i*)ga, (AS3 v4i*)&sA[l_row * LDST + l_k8], 0, 0);
    __builtin_amdgcn_global_load_async_to_lds_b128(
        (AS1 v4i*)gb, (AS3 v4i*)&sB[l_row * LDST + l_k8], 0, 0);
    __builtin_amdgcn_s_wait_asynccnt(0);
#else
    *(uint4*)&sA[l_row * LDST + l_k8] = *(const uint4*)ga;
    *(uint4*)&sB[l_row * LDST + l_k8] = *(const uint4*)gb;
#endif
    __syncthreads();

    bf16x16 a = frag_a(sA, LDST, wm, lane, 0);
#pragma unroll
    for (int nt = 0; nt < 2; ++nt) {
      bf16x16 b = frag_b(sB, LDST, wn + nt * 16, lane, 0);
      acc[nt] = wmma_bf16(a, b, acc[nt]);
    }
    __syncthreads();
  }

  const int mhalf = (lane & 16) ? 8 : 0;
#pragma unroll
  for (int nt = 0; nt < 2; ++nt)
#pragma unroll
    for (int r = 0; r < 8; ++r) {
      const int d = wm + r + mhalf;
      const int e = wn + nt * 16 + (lane & 15);
      atomicAdd(&ctx[(size_t)(h * 64 + d) * 64 + e], acc[nt][r]);
    }
}

// ---- out GEMM: attn[h][e][p] = sum_d ctx[h][d][e] * qb[h][d][p] ------------
__global__ void __launch_bounds__(256) out_gemm_k(
    const float* __restrict__ ctx, const bf16* __restrict__ qb,
    bf16* __restrict__ attnb)
{
  __shared__ bf16 sA[64 * LDO];    // A[e][d]  (ctx transposed, bf16)
  __shared__ bf16 sB[128 * LDO];   // B[p][d]
  const int h    = blockIdx.y;
  const int pblk = blockIdx.x * 128;
  const int t    = threadIdx.x;
  const int lane = t & 31;
  const int wid  = t >> 5;
  const int wm   = (wid >> 2) * 32;   // wave tile 32(e) x 32(p)
  const int wn   = (wid & 3) * 32;

#pragma unroll
  for (int i = 0; i < 16; ++i) {       // ctx head -> LDS, transposed
    const int idx = t * 16 + i;
    const int d = idx >> 6, e = idx & 63;
    sA[e * LDO + d] = (bf16)ctx[(size_t)(h * 64 + d) * 64 + e];
  }
  {
    // two adjacent d-rows per thread -> paired 4B LDS stores
    const int d2 = (t & 31) * 2;       // 0..62
    const int n0 = (t >> 5) * 16;      // 0..112
    const bf16* gq0 = qb + (size_t)(h * 64 + d2) * NPIX + pblk + n0;
    const bf16* gq1 = gq0 + NPIX;
    bf16 r0[16], r1[16];
    *(uint4*)&r0[0] = ((const uint4*)gq0)[0];
    *(uint4*)&r0[8] = ((const uint4*)gq0)[1];
    *(uint4*)&r1[0] = ((const uint4*)gq1)[0];
    *(uint4*)&r1[8] = ((const uint4*)gq1)[1];
#pragma unroll
    for (int i = 0; i < 16; ++i) {
      bf16x2 pr; pr[0] = r0[i]; pr[1] = r1[i];
      *(bf16x2*)&sB[(n0 + i) * LDO + d2] = pr;
    }
  }
  __syncthreads();

  f32x8 acc[2][2];
#pragma unroll
  for (int mt = 0; mt < 2; ++mt)
#pragma unroll
    for (int nt = 0; nt < 2; ++nt)
#pragma unroll
      for (int r = 0; r < 8; ++r) acc[mt][nt][r] = 0.0f;

#pragma unroll
  for (int kk = 0; kk < 64; kk += 32) {
    bf16x16 af[2];
#pragma unroll
    for (int mt = 0; mt < 2; ++mt) af[mt] = frag_a(sA, LDO, wm + mt * 16, lane, kk);
#pragma unroll
    for (int nt = 0; nt < 2; ++nt) {
      bf16x16 b = frag_b(sB, LDO, wn + nt * 16, lane, kk);
#pragma unroll
      for (int mt = 0; mt < 2; ++mt) acc[mt][nt] = wmma_bf16(af[mt], b, acc[mt][nt]);
    }
  }

  const int mhalf = (lane & 16) ? 8 : 0;
#pragma unroll
  for (int mt = 0; mt < 2; ++mt)
#pragma unroll
    for (int nt = 0; nt < 2; ++nt)
#pragma unroll
      for (int r = 0; r < 8; ++r) {
        const int e = wm + mt * 16 + r + mhalf;
        const int p = pblk + wn + nt * 16 + (lane & 15);
        attnb[(size_t)(h * 64 + e) * NPIX + p] = (bf16)acc[mt][nt][r];
      }
}

// ---- elementwise / reduction kernels ---------------------------------------

__global__ void convert_bf16_k(const float* __restrict__ s, bf16* __restrict__ d, int n) {
  const int i = blockIdx.x * 256 + threadIdx.x;
  if (i < n) d[i] = (bf16)s[i];
}

__global__ void rmsnorm_cast_k(const float* __restrict__ x, const float* __restrict__ g,
                               bf16* __restrict__ outb) {
  const int p = blockIdx.x * 256 + threadIdx.x;
  float s = 0.f;
  for (int c = 0; c < CDIM; ++c) { float v = x[(size_t)c * NPIX + p]; s += v * v; }
  const float rinv = 22.62741699796952f * rsqrtf(s);   // sqrt(512)/||x||
  for (int c = 0; c < CDIM; ++c)
    outb[(size_t)c * NPIX + p] = (bf16)(x[(size_t)c * NPIX + p] * rinv * (g[c] + 1.f));
}

__global__ void rmsnorm_out_k(const float* __restrict__ y, const float* __restrict__ g,
                              float* __restrict__ out) {
  const int p = blockIdx.x * 256 + threadIdx.x;
  float s = 0.f;
  for (int c = 0; c < CDIM; ++c) { float v = y[(size_t)c * NPIX + p]; s += v * v; }
  const float rinv = 22.62741699796952f * rsqrtf(s);
  for (int c = 0; c < CDIM; ++c)
    out[(size_t)c * NPIX + p] = y[(size_t)c * NPIX + p] * rinv * (g[c] + 1.f);
}

// q softmax over d (64), times DIM_HEAD^-0.5
__global__ void qsoft_k(const bf16* __restrict__ qkvb, bf16* __restrict__ qb) {
  const int p = blockIdx.x * 256 + threadIdx.x;
  const int h = blockIdx.y;
  const bf16* q = qkvb + (size_t)h * 64 * NPIX + p;
  float m = -1e30f;
  for (int d = 0; d < DHEAD; ++d) m = fmaxf(m, (float)q[(size_t)d * NPIX]);
  float s = 0.f;
  for (int d = 0; d < DHEAD; ++d) s += __expf((float)q[(size_t)d * NPIX] - m);
  const float inv = 0.125f / s;  // * 64^-0.5
  for (int d = 0; d < DHEAD; ++d)
    qb[(size_t)(h * 64 + d) * NPIX + p] = (bf16)(__expf((float)q[(size_t)d * NPIX] - m) * inv);
}

// k row stats over 16384 pixels + 4 mem-kv columns
__global__ void kstat_k(const bf16* __restrict__ qkvb, const float* __restrict__ memkv,
                        float* __restrict__ rowmax, float* __restrict__ rowsum) {
  __shared__ float red[256];
  const int r = blockIdx.x;        // 0..511  (= h*64+d)
  const int t = threadIdx.x;
  const bf16* krow = qkvb + (size_t)(512 + r) * NPIX;
  float m = -1e30f;
  for (int p = t; p < NPIX; p += 256) m = fmaxf(m, (float)krow[p]);
  for (int j = 0; j < 4; ++j) m = fmaxf(m, memkv[r * 4 + j]);
  red[t] = m; __syncthreads();
  for (int s = 128; s > 0; s >>= 1) { if (t < s) red[t] = fmaxf(red[t], red[t + s]); __syncthreads(); }
  const float rm = red[0]; __syncthreads();
  float s = 0.f;
  for (int p = t; p < NPIX; p += 256) s += __expf((float)krow[p] - rm);
  red[t] = s; __syncthreads();
  for (int st = 128; st > 0; st >>= 1) { if (t < st) red[t] += red[t + st]; __syncthreads(); }
  if (t == 0) {
    float tot = red[0];
    for (int j = 0; j < 4; ++j) tot += __expf(memkv[r * 4 + j] - rm);
    rowmax[r] = rm; rowsum[r] = tot;
  }
}

// write softmaxed k and raw v as bf16
__global__ void kvwrite_k(const bf16* __restrict__ qkvb,
                          const float* __restrict__ rowmax, const float* __restrict__ rowsum,
                          bf16* __restrict__ kbuf, bf16* __restrict__ vbuf) {
  const int p = blockIdx.x * 256 + threadIdx.x;
  const int r = blockIdx.y;
  const float rm = rowmax[r], inv = 1.f / rowsum[r];
  const float kv = (float)qkvb[(size_t)(512 + r) * NPIX + p];
  kbuf[(size_t)r * NPIX + p] = (bf16)(__expf(kv - rm) * inv);
  vbuf[(size_t)r * NPIX + p] = qkvb[(size_t)(1024 + r) * NPIX + p];
}

// initialize ctx with mem-kv contribution (4 columns, exact f32)
__global__ void ctxinit_k(const float* __restrict__ memkv,
                          const float* __restrict__ rowmax, const float* __restrict__ rowsum,
                          float* __restrict__ ctx) {
  const int idx = blockIdx.x * 256 + threadIdx.x;   // 8*64*64
  const int h = idx >> 12, d = (idx >> 6) & 63, e = idx & 63;
  const int r = h * 64 + d;
  const float rm = rowmax[r], inv = 1.f / rowsum[r];
  float s = 0.f;
  for (int j = 0; j < 4; ++j)
    s += __expf(memkv[r * 4 + j] - rm) * inv * memkv[2048 + (h * 64 + e) * 4 + j];
  ctx[(size_t)r * 64 + e] = s;
}

// ---------------------------------------------------------------------------

extern "C" void kernel_launch(void* const* d_in, const int* in_sizes, int n_in,
                              void* d_out, int out_size, void* d_ws, size_t ws_size,
                              hipStream_t stream) {
  const float* x      = (const float*)d_in[0];
  const float* g1     = (const float*)d_in[1];
  const float* qkv_w  = (const float*)d_in[2];
  const float* memkv  = (const float*)d_in[3];
  const float* conv_w = (const float*)d_in[4];
  const float* conv_b = (const float*)d_in[5];
  const float* g2     = (const float*)d_in[6];
  float* out = (float*)d_out;
  char*  ws  = (char*)d_ws;

  // workspace layout (bytes) — dead buffers are reused
  const size_t OFF_WQKV  = 0;                               // 1536*512*2
  const size_t OFF_WCONV = 1572864;                         // 512*512*2
  const size_t OFF_XNB   = 2097152;                         // 512*16384*2 (later: qb)
  const size_t OFF_QKVB  = 18874368;                        // 1536*16384*2
  const size_t OFF_ATTNB = OFF_QKVB;                        // reuse (qkv dead)
  const size_t OFF_Y     = OFF_QKVB + 16777216;             // reuse (f32, 32MB)
  const size_t OFF_KB    = 69206016;                        // 512*16384*2
  const size_t OFF_VB    = 85983232;                        // 512*16384*2
  const size_t OFF_RMAX  = 102760448;
  const size_t OFF_RSUM  = OFF_RMAX + 2048;
  const size_t OFF_CTX   = OFF_RSUM + 2048;                 // 8*64*64*4

  bf16*  wqkvb  = (bf16*)(ws + OFF_WQKV);
  bf16*  wconvb = (bf16*)(ws + OFF_WCONV);
  bf16*  xnb    = (bf16*)(ws + OFF_XNB);
  bf16*  qb     = (bf16*)(ws + OFF_XNB);
  bf16*  qkvb   = (bf16*)(ws + OFF_QKVB);
  bf16*  attnb  = (bf16*)(ws + OFF_ATTNB);
  float* yv     = (float*)(ws + OFF_Y);
  bf16*  kbuf   = (bf16*)(ws + OFF_KB);
  bf16*  vbuf   = (bf16*)(ws + OFF_VB);
  float* rmax   = (float*)(ws + OFF_RMAX);
  float* rsum   = (float*)(ws + OFF_RSUM);
  float* ctx    = (float*)(ws + OFF_CTX);

  convert_bf16_k<<<(1536 * 512 + 255) / 256, 256, 0, stream>>>(qkv_w, wqkvb, 1536 * 512);
  convert_bf16_k<<<(512 * 512 + 255) / 256, 256, 0, stream>>>(conv_w, wconvb, 512 * 512);
  rmsnorm_cast_k<<<NPIX / 256, 256, 0, stream>>>(x, g1, xnb);

  // qkv = Wqkv * xn : 1536 x 16384 x 512  -> bf16
  gemm128_bf16<<<dim3(NPIX / 128, 1536 / 128), 256, 0, stream>>>(
      wqkvb, xnb, 1536, NPIX, CDIM, nullptr, nullptr, qkvb);

  qsoft_k<<<dim3(NPIX / 256, HEADS), 256, 0, stream>>>(qkvb, qb);
  kstat_k<<<512, 256, 0, stream>>>(qkvb, memkv, rmax, rsum);
  kvwrite_k<<<dim3(NPIX / 256, 512), 256, 0, stream>>>(qkvb, rmax, rsum, kbuf, vbuf);
  ctxinit_k<<<(8 * 64 * 64) / 256, 256, 0, stream>>>(memkv, rmax, rsum, ctx);

  // ctx += k_soft * v^T : per head 64 x 64 x 16384, split-K (16 chunks)
  ctx_gemm_k<<<dim3(16, HEADS), 256, 0, stream>>>(kbuf, vbuf, ctx);

  // attn = ctx^T * q : per head 64 x 16384 x 64
  out_gemm_k<<<dim3(NPIX / 128, HEADS), 256, 0, stream>>>(ctx, qb, attnb);

  // y = Wconv * attn + b : 512 x 16384 x 512 -> f32
  gemm128_bf16<<<dim3(NPIX / 128, 512 / 128), 256, 0, stream>>>(
      wconvb, attnb, 512, NPIX, CDIM, yv, conv_b, nullptr);

  rmsnorm_out_k<<<NPIX / 256, 256, 0, stream>>>(yv, g2, out);
}
